// AttentionBlock_52390011077248
// MI455X (gfx1250) — compile-verified, matrix operand order checked by reading, not run on hardware
//
#include <hip/hip_runtime.h>
#include <hip/hip_bf16.h>

// Problem constants (match reference)
#define BATCH 4
#define CDIM 512
#define SEQ 4096          // 64*64
#define NGROUPS 32
#define CPG 16            // CDIM / NGROUPS
#define GN_EPS 1e-6f

typedef __attribute__((ext_vector_type(16))) __bf16 v16bf;
typedef __attribute__((ext_vector_type(8)))  __bf16 v8bf;
typedef __attribute__((ext_vector_type(8)))  float  v8f;
typedef __attribute__((ext_vector_type(4)))  unsigned int u32x4;
typedef __attribute__((ext_vector_type(8)))  int i32x8;
typedef __attribute__((ext_vector_type(4)))  int i32x4;

// Load one 16-bit WMMA operand fragment for this lane.
// Per CDNA5 ISA layout: lanes 0-15 hold k=[0..7]∪[16..23], lanes 16-31 hold
// k=[8..15]∪[24..31]. rowptr = start of this lane's row at current k offset.
static __device__ __forceinline__ v16bf frag_ld(const __bf16* rowptr, int lane) {
  const int kb = (lane >> 4) * 8;
  v8bf a = *reinterpret_cast<const v8bf*>(rowptr + kb);
  v8bf b = *reinterpret_cast<const v8bf*>(rowptr + kb + 16);
  v16bf r;
#pragma unroll
  for (int i = 0; i < 8; ++i) { r[i] = a[i]; r[i + 8] = b[i]; }
  return r;
}

// Raw LDS byte address (flat address truncates to LDS offset per aperture rules)
static __device__ __forceinline__ unsigned lds_addr32(const void* p) {
  return (unsigned)(unsigned long long)p;
}

// TDM: async 2-D bf16 tile load Global -> LDS via Tensor Data Mover (D# per ISA §8).
// tile_d0 = contiguous elements per row, tile_d1 = rows, stride0 = row stride (elems).
static __device__ __forceinline__ void tdm_load_2d(unsigned lds_addr, const void* gaddr,
                                                   unsigned tile_d0, unsigned tile_d1,
                                                   unsigned tensor_d0, unsigned tensor_d1,
                                                   unsigned stride0) {
  const unsigned long long ga = (unsigned long long)gaddr;
  u32x4 g0;
  g0[0] = 1u;                                  // count=1 (valid user descriptor)
  g0[1] = lds_addr;                            // lds_addr bytes
  g0[2] = (unsigned)ga;                        // global_addr[31:0]
  g0[3] = (unsigned)(ga >> 32) | (2u << 30);   // global_addr[56:32] | type=2 (image)
  i32x8 g1;
  g1[0] = 0x00010000;                          // workgroup_mask=0, data_size=1 (2B)
  g1[1] = (int)(tensor_d0 << 16);              // tensor_dim0[15:0] @ bits 63:48
  g1[2] = (int)((tensor_d0 >> 16) | (tensor_d1 << 16));   // dim0 hi | dim1 lo
  g1[3] = (int)(((tensor_d1 >> 16) & 0xFFFFu) | (tile_d0 << 16));  // dim1 hi | tile0
  g1[4] = (int)(tile_d1 & 0xFFFFu);            // tile_dim1 | tile_dim2=0
  g1[5] = (int)stride0;                        // tensor_dim0_stride[31:0]
  g1[6] = 0;                                   // stride0 hi | dim1_stride lo
  g1[7] = 0;                                   // dim1_stride hi
  i32x4 z4 = {0, 0, 0, 0};
#if defined(__clang_major__) && (__clang_major__ >= 23)
  i32x8 z8 = {0, 0, 0, 0, 0, 0, 0, 0};
  __builtin_amdgcn_tensor_load_to_lds(g0, g1, z4, z4, z8, 0);
#else
  __builtin_amdgcn_tensor_load_to_lds(g0, g1, z4, z4, 0);
#endif
}

// ---------------------------------------------------------------------------
// 1) GroupNorm: (B,C,H,W) fp32 -> normalized (B,S,C) bf16
// ---------------------------------------------------------------------------
__global__ __launch_bounds__(256) void gn_kernel(const float* __restrict__ x,
                                                 const float* __restrict__ gamma,
                                                 const float* __restrict__ beta,
                                                 __bf16* __restrict__ xn) {
  __shared__ float red[256];
  __shared__ float red2[256];
  const int blk = blockIdx.x;
  const int b = blk / NGROUPS, g = blk % NGROUPS;
  const float* xg = x + ((size_t)b * CDIM + (size_t)g * CPG) * SEQ;
  const int tid = threadIdx.x;
  const int N = CPG * SEQ;  // 65536

  float sum = 0.f, sumsq = 0.f;
  for (int j = tid; j < N; j += 256) {
    float v = xg[j];
    sum += v;
    sumsq += v * v;
  }
  red[tid] = sum;
  red2[tid] = sumsq;
  __syncthreads();
  for (int off = 128; off > 0; off >>= 1) {
    if (tid < off) { red[tid] += red[tid + off]; red2[tid] += red2[tid + off]; }
    __syncthreads();
  }
  const float mean = red[0] / (float)N;
  const float var = red2[0] / (float)N - mean * mean;
  const float rinv = rsqrtf(var + GN_EPS);

  for (int j = tid; j < N; j += 256) {
    const int c = g * CPG + (j >> 12);  // j / 4096
    const int s = j & (SEQ - 1);
    const float v = (xg[j] - mean) * rinv * gamma[c] + beta[c];
    xn[((size_t)b * SEQ + s) * CDIM + c] = (__bf16)v;
  }
}

// ---------------------------------------------------------------------------
// 2) fp32 -> bf16 (optionally scaled) weight conversion; fp32 bias scaling
// ---------------------------------------------------------------------------
__global__ void cvt_scale_kernel(const float* __restrict__ in, __bf16* __restrict__ out,
                                 int n, float s) {
  const int i = blockIdx.x * 256 + threadIdx.x;
  if (i < n) out[i] = (__bf16)(in[i] * s);
}

__global__ void scalef_kernel(const float* __restrict__ in, float* __restrict__ out,
                              int n, float s) {
  const int i = blockIdx.x * 256 + threadIdx.x;
  if (i < n) out[i] = in[i] * s;
}

// ---------------------------------------------------------------------------
// 3) WMMA GEMM: Y[M,512] = X[M,512] @ W^T + bias, bf16 out.
//    Block tile 128x128, 8 waves (4M x 2N), wave tile 32x64.
//    X tile staged by the Tensor Data Mover, double-buffered on TENSORcnt.
// ---------------------------------------------------------------------------
#define BM 128
#define BN 128
#define KC 64

__global__ __launch_bounds__(256, 1) void gemm_qkv(const __bf16* __restrict__ X,
                                                   const __bf16* __restrict__ W,
                                                   const float* __restrict__ bias,
                                                   __bf16* __restrict__ Y) {
  __shared__ __bf16 xs[2][BM][KC];  // 2 x 16 KB
  const int tid = threadIdx.x;
  const int lane = tid & 31;
  const int w = tid >> 5;
  const int mblk = blockIdx.x * BM;
  const int nblk = blockIdx.y * BN;
  const int wm = (w & 3) * 32;
  const int wn = (w >> 2) * 64;
  const __bf16* Xblk = X + (size_t)mblk * CDIM;

  v8f acc[2][4];
#pragma unroll
  for (int i = 0; i < 2; ++i)
#pragma unroll
    for (int j = 0; j < 4; ++j)
#pragma unroll
      for (int r = 0; r < 8; ++r) acc[i][j][r] = 0.f;

  // Prime the pipeline: TDM load of first X tile (wave 0 issues).
  if (w == 0)
    tdm_load_2d(lds_addr32(&xs[0][0][0]), Xblk, KC, BM, CDIM, BATCH * SEQ, CDIM);

  int buf = 0;
  for (int kc = 0; kc < CDIM; kc += KC) {
    if (w == 0) __builtin_amdgcn_s_wait_tensorcnt(0);
    __syncthreads();
    if (w == 0 && kc + KC < CDIM)
      tdm_load_2d(lds_addr32(&xs[buf ^ 1][0][0]), Xblk + kc + KC, KC, BM, CDIM,
                  BATCH * SEQ, CDIM);

#pragma unroll
    for (int ks = 0; ks < KC; ks += 32) {
      v16bf bfrag[4];
#pragma unroll
      for (int nt = 0; nt < 4; ++nt) {
        const int n = nblk + wn + nt * 16 + (lane & 15);
        bfrag[nt] = frag_ld(W + (size_t)n * CDIM + kc + ks, lane);
      }
#pragma unroll
      for (int mt = 0; mt < 2; ++mt) {
        const int mrow = wm + mt * 16 + (lane & 15);
        v16bf afrag = frag_ld(&xs[buf][mrow][ks], lane);
#pragma unroll
        for (int nt = 0; nt < 4; ++nt) {
          acc[mt][nt] = __builtin_amdgcn_wmma_f32_16x16x32_bf16(
              false, afrag, false, bfrag[nt], (short)0, acc[mt][nt], false, false);
        }
      }
    }
    __syncthreads();
    buf ^= 1;
  }

#pragma unroll
  for (int mt = 0; mt < 2; ++mt) {
#pragma unroll
    for (int nt = 0; nt < 4; ++nt) {
      const int n = nblk + wn + nt * 16 + (lane & 15);
      const float bv = bias[n];
#pragma unroll
      for (int r = 0; r < 8; ++r) {
        const int m = mblk + wm + mt * 16 + r + ((lane >> 4) * 8);
        Y[(size_t)m * CDIM + n] = (__bf16)(acc[mt][nt][r] + bv);
      }
    }
  }
}

// ---------------------------------------------------------------------------
// 4) Flash attention: 32 queries per block, 64-key tiles, online softmax.
//    Q pre-scaled (1/sqrt(C) folded into wq/bq); Q/K fragments straight from
//    global (L2-resident); V staged into LDS transposed in 32-key chunks so
//    P@V B-fragments are contiguous ds_load_b128 pairs.
// ---------------------------------------------------------------------------
__global__ __launch_bounds__(256, 1) void attn_kernel(const __bf16* __restrict__ Q,
                                                      const __bf16* __restrict__ K,
                                                      const __bf16* __restrict__ V,
                                                      __bf16* __restrict__ O) {
  __shared__ float sc[32][64];      // 8 KB scores
  __shared__ __bf16 ps[32][64];     // 4 KB probabilities
  __shared__ __bf16 vs[CDIM][32];   // 32 KB transposed V chunk [c][key]
  __shared__ float mrow[32], lrow[32], crow[32];

  const int tid = threadIdx.x;
  const int lane = tid & 31;
  const int w = tid >> 5;
  const int b = blockIdx.y;
  const int qbase = blockIdx.x * 32;
  const __bf16* Qb = Q + ((size_t)b * SEQ + qbase) * CDIM;
  const __bf16* Kb = K + (size_t)b * SEQ * CDIM;
  const __bf16* Vb = V + (size_t)b * SEQ * CDIM;

  if (tid < 32) { mrow[tid] = -1e30f; lrow[tid] = 0.f; }
  __syncthreads();

  const int mO = (w & 1) * 16;    // query row slab
  const int nO = (w >> 1) * 128;  // output channel slab
  const int nS = (w >> 1) * 16;   // key column slab for scores

  v8f oacc[8];
#pragma unroll
  for (int i = 0; i < 8; ++i)
#pragma unroll
    for (int r = 0; r < 8; ++r) oacc[i][r] = 0.f;

  for (int kt = 0; kt < SEQ; kt += 64) {
    if (kt + 64 < SEQ) {
      __builtin_prefetch(Kb + (size_t)(kt + 64 + (tid & 63)) * CDIM + (tid >> 6) * 32, 0, 0);
      __builtin_prefetch(Vb + (size_t)(kt + 64 + (tid & 63)) * CDIM + (tid >> 6) * 32, 0, 0);
    }

    // ---- scores: 16x16 tile per wave, full K=512 contraction
    v8f sacc;
#pragma unroll
    for (int r = 0; r < 8; ++r) sacc[r] = 0.f;
#pragma unroll
    for (int ks = 0; ks < CDIM; ks += 32) {
      v16bf af = frag_ld(Qb + (size_t)(mO + (lane & 15)) * CDIM + ks, lane);
      v16bf bfv = frag_ld(Kb + (size_t)(kt + nS + (lane & 15)) * CDIM + ks, lane);
      sacc = __builtin_amdgcn_wmma_f32_16x16x32_bf16(false, af, false, bfv, (short)0,
                                                     sacc, false, false);
    }
#pragma unroll
    for (int r = 0; r < 8; ++r)
      sc[mO + r + ((lane >> 4) * 8)][nS + (lane & 15)] = sacc[r];
    __syncthreads();

    // ---- online softmax (one row per thread of wave 0)
    if (tid < 32) {
      const int row = tid;
      const float mold = mrow[row];
      float mt = mold;
      for (int j = 0; j < 64; ++j) mt = fmaxf(mt, sc[row][j]);
      const float corr = __expf(mold - mt);
      float lsum = lrow[row] * corr;
      for (int j = 0; j < 64; ++j) {
        const float p = __expf(sc[row][j] - mt);
        ps[row][j] = (__bf16)p;
        lsum += p;
      }
      mrow[row] = mt;
      lrow[row] = lsum;
      crow[row] = corr;
    }
    __syncthreads();

    // ---- rescale running accumulators by the max-shift correction
#pragma unroll
    for (int i = 0; i < 8; ++i)
#pragma unroll
      for (int r = 0; r < 8; ++r) oacc[i][r] *= crow[mO + r + ((lane >> 4) * 8)];

    // ---- P @ V in two 32-key chunks (V staged transposed in LDS)
#pragma unroll 1
    for (int ck = 0; ck < 2; ++ck) {
      // stage: 32 keys x 512 channels, coalesced 16B reads, transposed writes
#pragma unroll
      for (int t = 0; t < 8; ++t) {
        const int slot = t * 256 + tid;   // 2048 slots of 8 elements
        const int key = slot >> 6;        // 0..31
        const int c8 = (slot & 63) * 8;   // 0..504
        v8bf vv = *reinterpret_cast<const v8bf*>(
            Vb + (size_t)(kt + ck * 32 + key) * CDIM + c8);
#pragma unroll
        for (int e = 0; e < 8; ++e) vs[c8 + e][key] = vv[e];
      }
      __syncthreads();

      v16bf af = frag_ld(&ps[mO + (lane & 15)][ck * 32], lane);
#pragma unroll
      for (int i = 0; i < 8; ++i) {
        const int cc = nO + i * 16 + (lane & 15);
        v16bf bfv = frag_ld(&vs[cc][0], lane);
        oacc[i] = __builtin_amdgcn_wmma_f32_16x16x32_bf16(false, af, false, bfv, (short)0,
                                                          oacc[i], false, false);
      }
      __syncthreads();
    }
  }

  // ---- final 1/l normalization and store
#pragma unroll
  for (int i = 0; i < 8; ++i) {
    const int cc = nO + i * 16 + (lane & 15);
#pragma unroll
    for (int r = 0; r < 8; ++r) {
      const int row = mO + r + ((lane >> 4) * 8);
      const float val = oacc[i][r] / lrow[row];
      O[((size_t)b * SEQ + qbase + row) * CDIM + cc] = (__bf16)val;
    }
  }
}

// ---------------------------------------------------------------------------
// 5) Output projection GEMM + bias + residual + (B,S,C)->(B,C,H,W) transpose
// ---------------------------------------------------------------------------
__global__ __launch_bounds__(256, 1) void gemm_out(const __bf16* __restrict__ X,
                                                   const __bf16* __restrict__ W,
                                                   const float* __restrict__ bias,
                                                   const float* __restrict__ resid,
                                                   float* __restrict__ out) {
  __shared__ __bf16 xs[2][BM][KC];
  const int tid = threadIdx.x;
  const int lane = tid & 31;
  const int w = tid >> 5;
  const int mblk = blockIdx.x * BM;
  const int nblk = blockIdx.y * BN;
  const int wm = (w & 3) * 32;
  const int wn = (w >> 2) * 64;
  const __bf16* Xblk = X + (size_t)mblk * CDIM;

  v8f acc[2][4];
#pragma unroll
  for (int i = 0; i < 2; ++i)
#pragma unroll
    for (int j = 0; j < 4; ++j)
#pragma unroll
      for (int r = 0; r < 8; ++r) acc[i][j][r] = 0.f;

  if (w == 0)
    tdm_load_2d(lds_addr32(&xs[0][0][0]), Xblk, KC, BM, CDIM, BATCH * SEQ, CDIM);

  int buf = 0;
  for (int kc = 0; kc < CDIM; kc += KC) {
    if (w == 0) __builtin_amdgcn_s_wait_tensorcnt(0);
    __syncthreads();
    if (w == 0 && kc + KC < CDIM)
      tdm_load_2d(lds_addr32(&xs[buf ^ 1][0][0]), Xblk + kc + KC, KC, BM, CDIM,
                  BATCH * SEQ, CDIM);

#pragma unroll
    for (int ks = 0; ks < KC; ks += 32) {
      v16bf bfrag[4];
#pragma unroll
      for (int nt = 0; nt < 4; ++nt) {
        const int n = nblk + wn + nt * 16 + (lane & 15);
        bfrag[nt] = frag_ld(W + (size_t)n * CDIM + kc + ks, lane);
      }
#pragma unroll
      for (int mt = 0; mt < 2; ++mt) {
        const int mrow = wm + mt * 16 + (lane & 15);
        v16bf afrag = frag_ld(&xs[buf][mrow][ks], lane);
#pragma unroll
        for (int nt = 0; nt < 4; ++nt) {
          acc[mt][nt] = __builtin_amdgcn_wmma_f32_16x16x32_bf16(
              false, afrag, false, bfrag[nt], (short)0, acc[mt][nt], false, false);
        }
      }
    }
    __syncthreads();
    buf ^= 1;
  }

#pragma unroll
  for (int mt = 0; mt < 2; ++mt) {
#pragma unroll
    for (int nt = 0; nt < 4; ++nt) {
      const int n = nblk + wn + nt * 16 + (lane & 15);
      const float bv = bias[n];
#pragma unroll
      for (int r = 0; r < 8; ++r) {
        const int m = mblk + wm + mt * 16 + r + ((lane >> 4) * 8);
        const int bb = m >> 12;           // m / 4096
        const int s = m & (SEQ - 1);
        const size_t oi = ((size_t)bb * CDIM + n) * SEQ + s;
        out[oi] = acc[mt][nt][r] + bv + resid[oi];
      }
    }
  }
}

// ---------------------------------------------------------------------------
extern "C" void kernel_launch(void* const* d_in, const int* in_sizes, int n_in,
                              void* d_out, int out_size, void* d_ws, size_t ws_size,
                              hipStream_t stream) {
  (void)in_sizes; (void)n_in; (void)out_size; (void)ws_size;

  const float* hidden = (const float*)d_in[0];
  const float* gnw = (const float*)d_in[1];
  const float* gnb = (const float*)d_in[2];
  const float* wq = (const float*)d_in[3];
  const float* bq = (const float*)d_in[4];
  const float* wk = (const float*)d_in[5];
  const float* bk = (const float*)d_in[6];
  const float* wv = (const float*)d_in[7];
  const float* bv = (const float*)d_in[8];
  const float* wo = (const float*)d_in[9];
  const float* bo = (const float*)d_in[10];
  float* out = (float*)d_out;

  char* ws = (char*)d_ws;
  const size_t nBSC = (size_t)BATCH * SEQ * CDIM;
  __bf16* xn = (__bf16*)ws;  ws += nBSC * 2;
  __bf16* q  = (__bf16*)ws;  ws += nBSC * 2;
  __bf16* k  = (__bf16*)ws;  ws += nBSC * 2;
  __bf16* v  = (__bf16*)ws;  ws += nBSC * 2;
  __bf16* o  = (__bf16*)ws;  ws += nBSC * 2;
  __bf16* wqs = (__bf16*)ws; ws += (size_t)CDIM * CDIM * 2;
  __bf16* wkb = (__bf16*)ws; ws += (size_t)CDIM * CDIM * 2;
  __bf16* wvb = (__bf16*)ws; ws += (size_t)CDIM * CDIM * 2;
  __bf16* wob = (__bf16*)ws; ws += (size_t)CDIM * CDIM * 2;
  float* bqs = (float*)ws;   ws += (size_t)CDIM * 4;

  const float attn_scale = 0.044194173824159216f;  // 1/sqrt(512)
  const int wsz = CDIM * CDIM;
  cvt_scale_kernel<<<wsz / 256, 256, 0, stream>>>(wq, wqs, wsz, attn_scale);
  cvt_scale_kernel<<<wsz / 256, 256, 0, stream>>>(wk, wkb, wsz, 1.0f);
  cvt_scale_kernel<<<wsz / 256, 256, 0, stream>>>(wv, wvb, wsz, 1.0f);
  cvt_scale_kernel<<<wsz / 256, 256, 0, stream>>>(wo, wob, wsz, 1.0f);
  scalef_kernel<<<2, 256, 0, stream>>>(bq, bqs, CDIM, attn_scale);

  gn_kernel<<<BATCH * NGROUPS, 256, 0, stream>>>(hidden, gnw, gnb, xn);

  dim3 gg((BATCH * SEQ) / BM, CDIM / BN);  // (128, 4)
  gemm_qkv<<<gg, 256, 0, stream>>>(xn, wqs, bqs, q);
  gemm_qkv<<<gg, 256, 0, stream>>>(xn, wkb, bk, k);
  gemm_qkv<<<gg, 256, 0, stream>>>(xn, wvb, bv, v);

  dim3 ga(SEQ / 32, BATCH);  // (128, 4)
  attn_kernel<<<ga, 256, 0, stream>>>(q, k, v, o);

  gemm_out<<<gg, 256, 0, stream>>>(o, wob, bo, hidden, out);
}
